// DynamicConv2D_65558380806413
// MI455X (gfx1250) — compile-verified, matrix operand order checked by reading, not run on hardware
//
#include <hip/hip_runtime.h>
#include <hip/hip_bf16.h>

typedef __attribute__((ext_vector_type(16))) _Float16 v16h;
typedef __attribute__((ext_vector_type(8)))  float    v8f;

#define O_    128
#define C_    64
#define KH_   3
#define KW_   3
#define B_    16
#define H_    128
#define W_    128
#define DYN_H 384
#define DYN_W 192
#define RANK_ 16
#define CTX_  64
#define KTOT  (C_ * KH_ * KW_)   /* 576 */
#define XT_COLS 132              /* 128 + 2 halo + pad */

// --------------------------------------------------------------------------
// Kernel 0: pre-transpose mod_b[c][r][w] -> mbT[c][w][r] as f16 so the
// synthesis B-fragment (fixed w, r=0..15 per lane) is one contiguous 32B
// load instead of 16 wait-serialized strided b32 loads.
// --------------------------------------------------------------------------
__global__ __launch_bounds__(256) void pack_modb(
    const float* __restrict__ mod_b, _Float16* __restrict__ mbT)
{
  const int e = blockIdx.x * 256 + threadIdx.x;   // < 64*16*192
  const int w = e % DYN_W;                        // fastest -> coalesced read
  const int r = (e / DYN_W) % RANK_;
  const int c = e / (DYN_W * RANK_);
  mbT[((size_t)c * DYN_W + w) * RANK_ + r] = (_Float16)mod_b[e];
}

// --------------------------------------------------------------------------
// Kernel 1: context-conditioned low-rank weight synthesis.
// dw[b,h,w] = sum_{c,r} ctx[b,c] * mod_a[c,h,r] * mod_b[c,r,w] + bias_dyn
// Joint K index kappa = c*16 + r (K = 1024). One wave per 16x16 tile of dw.
// Each 32-wide K chunk covers 2 channels x rank 16: both fragments are
// 16 contiguous elements per lane (A: f32*scale->f16, B: packed f16).
// --------------------------------------------------------------------------
__global__ __launch_bounds__(256) void dyn_weight_synth(
    const float* __restrict__ ctx, const float* __restrict__ mod_a,
    const _Float16* __restrict__ mbT, const float* __restrict__ bias_dyn,
    float* __restrict__ dw)
{
  const int tid  = threadIdx.x;
  const int lane = tid & 31;
  const int sel  = lane >> 4;          // half-wave select
  const int ln   = lane & 15;
  const int wave = blockIdx.x * 8 + (tid >> 5);   // 4608 tiles total
  const int b  = wave / 288;
  const int t  = wave % 288;           // 24 x 12 tiles of 16x16
  const int h0 = (t / 12) * 16;
  const int w0 = (t % 12) * 16;

  v8f acc = {};
  #pragma unroll 4
  for (int kc = 0; kc < 32; ++kc) {    // K = 1024, 32 per WMMA
    const int   c     = kc * 2 + sel;  // channel for this half-wave
    const float scale = ctx[b * CTX_ + c];
    const float* ap = mod_a + ((size_t)c * DYN_H + (h0 + ln)) * RANK_;
    v16h a;
    #pragma unroll
    for (int i = 0; i < 16; ++i) a[i] = (_Float16)(ap[i] * scale);
    const v16h bm = *(const v16h*)(mbT + ((size_t)c * DYN_W + (w0 + ln)) * RANK_);
    acc = __builtin_amdgcn_wmma_f32_16x16x32_f16(false, a, false, bm,
                                                 (short)0, acc, false, false);
  }
  #pragma unroll
  for (int v = 0; v < 8; ++v) {
    const int row = h0 + v + 8 * sel;
    const int col = w0 + ln;
    dw[((size_t)b * DYN_H + row) * DYN_W + col] =
        acc[v] + bias_dyn[row * DYN_W + col];
  }
}

// --------------------------------------------------------------------------
// Kernel 2: apply dyn_index gather, pack conv weights as f16 in
// wf[b][o][k], k = (kh*3+kw)*64 + c  (k contiguous per output channel so
// conv A-fragments are single 32B loads).
// --------------------------------------------------------------------------
__global__ __launch_bounds__(256) void gather_pack(
    const float* __restrict__ dw, const int* __restrict__ dyn_index,
    _Float16* __restrict__ wf)
{
  const int e = blockIdx.x * 256 + threadIdx.x;   // < 16*128*576
  const int b = e / (O_ * KTOT);
  const int r = e % (O_ * KTOT);
  const int o = r / KTOT;
  const int k = r % KTOT;
  const int s = k >> 6;                // spatial tap 0..8
  const int c = k & 63;
  const int kh = s / 3, kw = s % 3;
  const int src = dyn_index[((o * C_ + c) * KH_ + kh) * KW_ + kw];
  wf[e] = (_Float16)dw[(size_t)b * (DYN_H * DYN_W) + src];
}

// --------------------------------------------------------------------------
// Kernel 3: implicit-GEMM 3x3 conv. One block per (b, h) output row.
// LDS stage: x[b, :, h-1..h+1, -1..128] as f16 in [dh][col][c] layout
// (c innermost -> B fragments are contiguous, 32B-aligned LDS reads).
// 8 waves: wave w owns output channels [16w, 16w+16) x all 128 pixels
// = 8 f32 WMMA accumulators; K = 576 in 18 steps of 32.
// --------------------------------------------------------------------------
__global__ __launch_bounds__(256) void conv_wmma(
    const float* __restrict__ x, const _Float16* __restrict__ wf,
    const float* __restrict__ bias_conv, float* __restrict__ y)
{
  __shared__ __align__(32) _Float16 xs[3 * XT_COLS * C_];   // ~50.7 KB

  const int b   = blockIdx.x >> 7;
  const int h   = blockIdx.x & 127;
  const int tid = threadIdx.x;

  // cooperative stage with zero-padded halo (coalesced along image columns)
  for (int e = tid; e < 3 * XT_COLS * C_; e += 256) {
    const int cl = e % XT_COLS;
    const int c  = (e / XT_COLS) % C_;
    const int dh = e / (XT_COLS * C_);
    const int hh = h + dh - 1;
    const int ww = cl - 1;
    float v = 0.f;
    if (hh >= 0 && hh < H_ && ww >= 0 && ww < W_)
      v = x[(((size_t)b * C_ + c) * H_ + hh) * W_ + ww];
    xs[((size_t)dh * XT_COLS + cl) * C_ + c] = (_Float16)v;
  }
  __syncthreads();

  const int lane = tid & 31;
  const int sel  = lane >> 4;
  const int ln   = lane & 15;
  const int o0   = (tid >> 5) * 16;

  v8f acc[8] = {};

  const _Float16* wrow = wf + ((size_t)b * O_ + (o0 + ln)) * KTOT;

  #pragma unroll 2
  for (int st = 0; st < 18; ++st) {
    const int s   = st >> 1;            // spatial tap
    const int c0  = (st & 1) * 32;      // channel half
    const int dh  = s / 3;
    const int dwx = s % 3;
    const v16h a = *(const v16h*)(wrow + st * 32 + sel * 16);
    #pragma unroll
    for (int j = 0; j < 8; ++j) {
      const int col = j * 16 + ln;      // output pixel column
      const v16h bm = *(const v16h*)&xs[((dh * XT_COLS + col + dwx) * C_)
                                        + c0 + sel * 16];
      acc[j] = __builtin_amdgcn_wmma_f32_16x16x32_f16(false, a, false, bm,
                                                      (short)0, acc[j],
                                                      false, false);
    }
  }

  #pragma unroll
  for (int j = 0; j < 8; ++j) {
    const int px = j * 16 + ln;
    #pragma unroll
    for (int v = 0; v < 8; ++v) {
      const int o = o0 + v + 8 * sel;
      y[(((size_t)b * O_ + o) * H_ + h) * W_ + px] = acc[j][v] + bias_conv[o];
    }
  }
}

// --------------------------------------------------------------------------
extern "C" void kernel_launch(void* const* d_in, const int* in_sizes, int n_in,
                              void* d_out, int out_size, void* d_ws, size_t ws_size,
                              hipStream_t stream) {
  const float* x        = (const float*)d_in[0];   // [16,64,128,128]
  const float* ctx      = (const float*)d_in[1];   // [16,64]
  const float* mod_a    = (const float*)d_in[2];   // [64,384,16]
  const float* mod_b    = (const float*)d_in[3];   // [64,16,192]
  const float* bias_dyn = (const float*)d_in[4];   // [1,384,192]
  const float* bias_cv  = (const float*)d_in[5];   // [128]
  const int*   dyn_idx  = (const int*)d_in[6];     // [128,64,3,3]
  float* out = (float*)d_out;                      // [16,128,128,128]

  char* ws = (char*)d_ws;
  float*    dw  = (float*)ws;                                     // 4.72 MB
  ws += (size_t)B_ * DYN_H * DYN_W * sizeof(float);
  _Float16* wfp = (_Float16*)ws;                                  // 2.36 MB
  ws += (size_t)B_ * O_ * KTOT * sizeof(_Float16);
  _Float16* mbT = (_Float16*)ws;                                  // 0.39 MB

  pack_modb       <<<768,  256, 0, stream>>>(mod_b, mbT);
  dyn_weight_synth<<<576,  256, 0, stream>>>(ctx, mod_a, mbT, bias_dyn, dw);
  gather_pack     <<<4608, 256, 0, stream>>>(dw, dyn_idx, wfp);
  conv_wmma       <<<2048, 256, 0, stream>>>(x, wfp, bias_cv, out);
}
